// JambaSparseMoeBlock_27736898797983
// MI455X (gfx1250) — matrix operand order repaired and checked by reading, not compiled, not run on hardware
//
#include <hip/hip_runtime.h>

#define HD 1024      // hidden size
#define ID 2816      // intermediate size
#define NE 8         // experts
#define NT 2048      // tokens (2*1024)
#define CHUNK 128    // intermediate-dim chunk per block iteration
#define GM 32        // tokens per block (two 16-row WMMA subtiles)
#define XP 8         // LDS row padding (bf16 elems = 16B) -> bank-conflict-free A reads

typedef __attribute__((ext_vector_type(16))) __bf16 v16bf;
typedef __attribute__((ext_vector_type(8)))  float  v8f;

union BFV {
  v16bf v;
  unsigned u[8];
  uint4 q[2];
};

// pack two floats to packed bf16 (truncate) with a single v_perm_b32:
// result = { lo16 = hi16(a), hi16 = hi16(b) }
__device__ __forceinline__ unsigned pack_bf16x2(float a, float b) {
  return __builtin_amdgcn_perm(__float_as_uint(b), __float_as_uint(a), 0x07060302u);
}

__device__ __forceinline__ unsigned short bf16_trunc(float f) {
  return (unsigned short)(__float_as_uint(f) >> 16);
}

__device__ __forceinline__ v8f wmma_bf16(v16bf a, v16bf b, v8f c) {
  // D(16x16 f32) = A(16x32 bf16) * B(32x16 bf16) + C
  return __builtin_amdgcn_wmma_f32_16x16x32_bf16(
      /*neg_a=*/false, a, /*neg_b=*/false, b,
      /*c_mod=*/(short)0, c, /*reuse_a=*/false, /*reuse_b=*/false);
}

// ---------------------------------------------------------------------------
// Kernel 1: router logits + softmax + top-2 + per-expert token list build,
// plus one-time fp32 -> bf16 conversion of the token's activation row.
// One 128-thread block per token.
// ---------------------------------------------------------------------------
__global__ __launch_bounds__(128)
void router_kernel(const float* __restrict__ x, const float* __restrict__ rw,
                   unsigned* __restrict__ xbf2 /* packed bf16 pairs */,
                   int* __restrict__ counts,
                   int* __restrict__ tok_id, float* __restrict__ tok_w) {
  const int t = blockIdx.x;
  const int tid = threadIdx.x;
  __shared__ float red[NE][128];

  float acc[NE];
#pragma unroll
  for (int e = 0; e < NE; e++) acc[e] = 0.f;

  const float* xr = x + (size_t)t * HD;
  for (int hh = tid * 2; hh < HD; hh += 256) {
    float a = xr[hh], b = xr[hh + 1];
    xbf2[((size_t)t * HD + hh) >> 1] = pack_bf16x2(a, b);
#pragma unroll
    for (int e = 0; e < NE; e++)
      acc[e] += a * rw[e * HD + hh] + b * rw[e * HD + hh + 1];
  }
#pragma unroll
  for (int e = 0; e < NE; e++) red[e][tid] = acc[e];
  __syncthreads();
  for (int s = 64; s > 0; s >>= 1) {
    if (tid < s) {
#pragma unroll
      for (int e = 0; e < NE; e++) red[e][tid] += red[e][tid + s];
    }
    __syncthreads();
  }

  if (tid == 0) {
    float p[NE];
    float mx = red[0][0];
#pragma unroll
    for (int e = 1; e < NE; e++) mx = fmaxf(mx, red[e][0]);
    float sum = 0.f;
#pragma unroll
    for (int e = 0; e < NE; e++) { p[e] = __expf(red[e][0] - mx); sum += p[e]; }
    float inv = 1.f / sum;
    int i0 = 0;
#pragma unroll
    for (int e = 1; e < NE; e++) if (p[e] > p[i0]) i0 = e;
    int i1 = (i0 == 0) ? 1 : 0;
#pragma unroll
    for (int e = 0; e < NE; e++) if (e != i0 && p[e] > p[i1]) i1 = e;
    // top-2 softmax probs, NOT renormalized (Jamba)
    int s0 = atomicAdd(&counts[i0], 1);
    tok_id[i0 * NT + s0] = t;
    tok_w[i0 * NT + s0]  = p[i0] * inv;
    int s1 = atomicAdd(&counts[i1], 1);
    tok_id[i1 * NT + s1] = t;
    tok_w[i1 * NT + s1]  = p[i1] * inv;
  }
}

// ---------------------------------------------------------------------------
// Kernel 2: fused sparse expert MLP. 32 gathered tokens per block (two 16-row
// WMMA subtiles sharing every weight (B) operand).
//   phase 1: wave w owns i-cols [w*16,w*16+16): gate/up GEMM over K=H=1024,
//            SiLU*up*routeweight -> bf16 tile in LDS.
//   phase 2: wave w owns H-cols [w*128,w*128+128): 16 f32 accumulator tiles
//            held in VGPRs across all 22 I-chunks.
// LDS rows padded by 16B so the 16 lanes of each 16B A-read tile all 64 banks.
// ---------------------------------------------------------------------------
__global__ __launch_bounds__(256)
void moe_mlp_kernel(const unsigned short* __restrict__ xbf,
                    const float* __restrict__ gate_w,
                    const float* __restrict__ up_w,
                    const float* __restrict__ down_w,
                    const int* __restrict__ counts,
                    const int* __restrict__ tok_id,
                    const float* __restrict__ tok_w,
                    float* __restrict__ y) {
  const int e    = blockIdx.y;
  const int grp  = blockIdx.x;
  const int cnt  = counts[e];
  const int base = grp * GM;
  if (base >= cnt) return;

  __shared__ alignas(16) unsigned short xs[GM][HD + XP];     // ~66 KB gathered x (bf16)
  __shared__ alignas(16) unsigned short hs[GM][CHUNK + XP];  // ~8.5 KB silu(g)*u (bf16)
  __shared__ int   ts[GM];
  __shared__ float wts[GM];

  const int tid  = threadIdx.x;
  const int wave = tid >> 5;
  const int lane = tid & 31;
  const int ln   = lane & 15;
  const int half = lane >> 4;

  if (tid < GM) {
    int idx  = base + tid;
    int cidx = idx < cnt ? idx : cnt - 1;
    ts[tid]  = tok_id[e * NT + cidx];
    wts[tid] = (idx < cnt) ? tok_w[e * NT + cidx] : 0.f;  // padded rows contribute 0
  }
  __syncthreads();

  // gather GM token rows (bf16) into LDS, 16B per transfer
  for (int v = tid; v < GM * (HD / 8); v += 256) {
    int m = v >> 7;       // HD/8 == 128
    int c = v & 127;
    ((uint4*)&xs[m][0])[c] = ((const uint4*)(xbf + (size_t)ts[m] * HD))[c];
  }
  __syncthreads();

  const float* Ge = gate_w + (size_t)e * ID * HD;
  const float* Ue = up_w   + (size_t)e * ID * HD;
  const float* De = down_w + (size_t)e * HD * ID;

  v8f ya[8], yb[8];
#pragma unroll
  for (int tI = 0; tI < 8; tI++)
#pragma unroll
    for (int r = 0; r < 8; r++) { ya[tI][r] = 0.f; yb[tI][r] = 0.f; }

  for (int ic = 0; ic < ID; ic += CHUNK) {
    // ---------------- phase 1: gate/up GEMM + SiLU for wave's i-tile ----------------
    const int i0 = ic + wave * 16;
    v8f g0, u0, g1, u1;
#pragma unroll
    for (int r = 0; r < 8; r++) { g0[r] = 0.f; u0[r] = 0.f; g1[r] = 0.f; u1[r] = 0.f; }

    const float* gr0 = Ge + (size_t)(i0 + ln) * HD + half * 16;
    const float* ur0 = Ue + (size_t)(i0 + ln) * HD + half * 16;
    for (int k0 = 0; k0 < HD; k0 += 32) {
      BFV A0, A1, Bg, Bu;
      // A (16x32 bf16): lane = token row; 16 consecutive K values per lane (k = 16*half + s)
      A0.q[0] = *(const uint4*)&xs[ln][k0 + half * 16];
      A0.q[1] = *(const uint4*)&xs[ln][k0 + half * 16 + 8];
      A1.q[0] = *(const uint4*)&xs[16 + ln][k0 + half * 16];
      A1.q[1] = *(const uint4*)&xs[16 + ln][k0 + half * 16 + 8];
      __builtin_prefetch(gr0 + k0 + 256, 0, 1);
      __builtin_prefetch(ur0 + k0 + 256, 0, 1);
      // B (32x16 bf16): lane = i col; 16 consecutive h values, packed by v_perm
#pragma unroll
      for (int q = 0; q < 4; q++) {
        float4 fg = ((const float4*)(gr0 + k0))[q];
        float4 fu = ((const float4*)(ur0 + k0))[q];
        Bg.u[q * 2 + 0] = pack_bf16x2(fg.x, fg.y);
        Bg.u[q * 2 + 1] = pack_bf16x2(fg.z, fg.w);
        Bu.u[q * 2 + 0] = pack_bf16x2(fu.x, fu.y);
        Bu.u[q * 2 + 1] = pack_bf16x2(fu.z, fu.w);
      }
      g0 = wmma_bf16(A0.v, Bg.v, g0);
      g1 = wmma_bf16(A1.v, Bg.v, g1);
      u0 = wmma_bf16(A0.v, Bu.v, u0);
      u1 = wmma_bf16(A1.v, Bu.v, u1);
    }
    // silu(g) * u * routing weight -> LDS (weight folded here; down proj is linear)
#pragma unroll
    for (int r = 0; r < 8; r++) {
      int m = r + half * 8;
      float ga = g0[r];
      float sa = ga * __builtin_amdgcn_rcpf(1.f + __expf(-ga));
      hs[m][wave * 16 + ln] = bf16_trunc(sa * u0[r] * wts[m]);
      float gb = g1[r];
      float sb = gb * __builtin_amdgcn_rcpf(1.f + __expf(-gb));
      hs[16 + m][wave * 16 + ln] = bf16_trunc(sb * u1[r] * wts[16 + m]);
    }
    __syncthreads();

    // ---------------- phase 2: down GEMM; wave owns H cols [wave*128, +128) ----------------
    const float* dr0 = De + (size_t)(wave * 128 + ln) * ID + ic + half * 16;
    for (int kk = 0; kk < CHUNK; kk += 32) {
      BFV A0, A1;
      A0.q[0] = *(const uint4*)&hs[ln][kk + half * 16];
      A0.q[1] = *(const uint4*)&hs[ln][kk + half * 16 + 8];
      A1.q[0] = *(const uint4*)&hs[16 + ln][kk + half * 16];
      A1.q[1] = *(const uint4*)&hs[16 + ln][kk + half * 16 + 8];
      __builtin_prefetch(dr0 + kk + 256, 0, 1);
#pragma unroll
      for (int tI = 0; tI < 8; tI++) {
        const float* dr = dr0 + (size_t)tI * 16 * ID + kk;
        BFV Bd;
#pragma unroll
        for (int q = 0; q < 4; q++) {
          float4 fd = ((const float4*)dr)[q];
          Bd.u[q * 2 + 0] = pack_bf16x2(fd.x, fd.y);
          Bd.u[q * 2 + 1] = pack_bf16x2(fd.z, fd.w);
        }
        ya[tI] = wmma_bf16(A0.v, Bd.v, ya[tI]);
        yb[tI] = wmma_bf16(A1.v, Bd.v, yb[tI]);
      }
    }
    __syncthreads();  // hs reused next chunk
  }

  // epilogue: scatter-accumulate into y (a token appears under 2 experts)
#pragma unroll
  for (int tI = 0; tI < 8; tI++) {
#pragma unroll
    for (int r = 0; r < 8; r++) {
      int m = r + half * 8;
      int col = wave * 128 + tI * 16 + ln;
      atomicAdd(&y[(size_t)ts[m] * HD + col], ya[tI][r]);
      atomicAdd(&y[(size_t)ts[16 + m] * HD + col], yb[tI][r]);
    }
  }
}

// ---------------------------------------------------------------------------
extern "C" void kernel_launch(void* const* d_in, const int* in_sizes, int n_in,
                              void* d_out, int out_size, void* d_ws, size_t ws_size,
                              hipStream_t stream) {
  const float* x  = (const float*)d_in[0];   // [2,1024,1024]
  const float* rw = (const float*)d_in[1];   // [8,1024]
  const float* gw = (const float*)d_in[2];   // [8,2816,1024]
  const float* uw = (const float*)d_in[3];   // [8,2816,1024]
  const float* dw = (const float*)d_in[4];   // [8,1024,2816]
  float* y = (float*)d_out;                  // [2,1024,1024] fp32

  char* ws = (char*)d_ws;
  int*      counts = (int*)ws;                                    // 8 ints (256B region)
  int*      tok_id = (int*)(ws + 256);                            // 8*2048 ints = 64KB
  float*    tok_w  = (float*)(ws + 256 + NE * NT * 4);            // 64KB
  unsigned* xbf2   = (unsigned*)(ws + 256 + 2 * NE * NT * 4);     // 2048*1024 bf16 = 4MB

  hipMemsetAsync(counts, 0, NE * sizeof(int), stream);
  hipMemsetAsync(y, 0, (size_t)out_size * sizeof(float), stream);

  router_kernel<<<NT, 128, 0, stream>>>(x, rw, xbf2, counts, tok_id, tok_w);

  dim3 grid(NT / GM, NE);  // 64 groups per expert (worst case: all tokens on one expert)
  moe_mlp_kernel<<<grid, 256, 0, stream>>>((const unsigned short*)xbf2, gw, uw, dw,
                                           counts, tok_id, tok_w, y);
}
